// GSAttention_21337397527393
// MI455X (gfx1250) — matrix operand order repaired and checked by reading, not compiled, not run on hardware
//
#include <hip/hip_runtime.h>
#include <hip/hip_bf16.h>

typedef __attribute__((ext_vector_type(16))) _Float16 v16h;
typedef __attribute__((ext_vector_type(8)))  _Float16 v8h;
typedef __attribute__((ext_vector_type(8)))  float    v8f;

#define DIMC   768
#define NHEAD  12
#define HD     64
#define NTOK   4096   // 64*64
#define NRED   1024   // 32*32
#define BATCH  2

__device__ __forceinline__ v16h cat8(v8h lo, v8h hi) {
  return __builtin_shufflevector(lo, hi, 0, 1, 2, 3, 4, 5, 6, 7,
                                         8, 9, 10, 11, 12, 13, 14, 15);
}

// ---------------------------------------------------------------------------
// f32 -> f16 convert (plain copy)
// ---------------------------------------------------------------------------
__global__ void cvt_f16_kernel(const float* __restrict__ in,
                               _Float16* __restrict__ out, int n) {
  const int i = blockIdx.x * blockDim.x + threadIdx.x;
  if (i < n) out[i] = (_Float16)in[i];
}

// sr_w permute-convert: in (O,C,2,2) -> out[o][khw*768 + c]
__global__ void cvt_srw_kernel(const float* __restrict__ in,
                               _Float16* __restrict__ out) {
  const int i = blockIdx.x * blockDim.x + threadIdx.x;
  if (i >= DIMC * DIMC * 4) return;
  const int o   = i / (DIMC * 4);
  const int r   = i - o * (DIMC * 4);
  const int khw = r / DIMC;
  const int c   = r - khw * DIMC;
  out[i] = (_Float16)in[(size_t)o * (DIMC * 4) + c * 4 + khw];
}

// ---------------------------------------------------------------------------
// f16 WMMA GEMM: C[m,n] = sum_k A[m,k]*W[n,k] (+bias[n]), C f32.
// 128 threads/block; block tile 64x64; each wave one 16x64 slice (4 WMMAs
// per k-step reusing a single A fragment). All fragment loads are b128.
// Requires M%64==0, N%64==0, K%32==0.
// ---------------------------------------------------------------------------
__global__ void gemm_h_kernel(const _Float16* __restrict__ A,
                              const _Float16* __restrict__ W,
                              const float* __restrict__ bias,
                              float* __restrict__ C,
                              int M, int N, int K) {
  const int ntile = N >> 6;
  const int tile  = blockIdx.x;
  const int wave  = threadIdx.x >> 5;
  const int lane  = threadIdx.x & 31;
  const int tm = ((tile / ntile) << 6) + (wave << 4);
  const int tn = (tile % ntile) << 6;
  const int abase = (lane < 16) ? 0 : 8;
  const int bbase = (lane < 16) ? 0 : 16;

  const _Float16* Arow = A + (size_t)(tm + (lane & 15)) * K;
  const _Float16* W0 = W + (size_t)(tn + (lane & 15)) * K + bbase;
  const _Float16* W1 = W0 + (size_t)16 * K;
  const _Float16* W2 = W0 + (size_t)32 * K;
  const _Float16* W3 = W0 + (size_t)48 * K;

  v8f acc[4] = {};
  for (int k0 = 0; k0 < K; k0 += 32) {
    const v16h a = *(const v16h*)(Arow + k0 + bbase);
    const v16h b0 = *(const v16h*)(W0 + k0);
    const v16h b1 = *(const v16h*)(W1 + k0);
    const v16h b2 = *(const v16h*)(W2 + k0);
    const v16h b3 = *(const v16h*)(W3 + k0);
    acc[0] = __builtin_amdgcn_wmma_f32_16x16x32_f16(false, a, false, b0, (short)0, acc[0], false, false);
    acc[1] = __builtin_amdgcn_wmma_f32_16x16x32_f16(false, a, false, b1, (short)0, acc[1], false, false);
    acc[2] = __builtin_amdgcn_wmma_f32_16x16x32_f16(false, a, false, b2, (short)0, acc[2], false, false);
    acc[3] = __builtin_amdgcn_wmma_f32_16x16x32_f16(false, a, false, b3, (short)0, acc[3], false, false);
  }

  const int r0 = (lane >> 4) << 3;
  const int cl = lane & 15;
#pragma unroll
  for (int t = 0; t < 4; ++t) {
    const int col = tn + (t << 4) + cl;
    const float bv = bias ? bias[col] : 0.0f;
#pragma unroll
    for (int r = 0; r < 8; ++r)
      C[(size_t)(tm + r0 + r) * N + col] = acc[t][r] + bv;
  }
}

// ---------------------------------------------------------------------------
// Stride-2 2x2 conv as GEMM with pixel-major K ordering (matches cvt_srw):
// k' = khw*768 + c, so every k-step's A chunks are contiguous f16 loads.
// M = B*1024, N = 768, K = 3072.
// ---------------------------------------------------------------------------
__global__ void conv_h_kernel(const _Float16* __restrict__ xh,
                              const _Float16* __restrict__ wh,
                              const float* __restrict__ bias,
                              float* __restrict__ out) {
  const int ntile = DIMC >> 6;           // 12
  const int tile  = blockIdx.x;
  const int wave  = threadIdx.x >> 5;
  const int lane  = threadIdx.x & 31;
  const int tm = ((tile / ntile) << 6) + (wave << 4);
  const int tn = (tile % ntile) << 6;
  const int abase = (lane < 16) ? 0 : 8;
  const int bbase = (lane < 16) ? 0 : 16;

  const int mrow = tm + (lane & 15);
  const int b  = mrow >> 10;
  const int p  = mrow & 1023;
  const int oi = p >> 5, oj = p & 31;
  int nin[4];
#pragma unroll
  for (int khw = 0; khw < 4; ++khw)
    nin[khw] = ((oi << 1) + (khw >> 1)) * 64 + ((oj << 1) + (khw & 1));

  const _Float16* W0 = wh + (size_t)(tn + (lane & 15)) * (DIMC * 4) + bbase;
  const _Float16* W1 = W0 + (size_t)16 * (DIMC * 4);
  const _Float16* W2 = W0 + (size_t)32 * (DIMC * 4);
  const _Float16* W3 = W0 + (size_t)48 * (DIMC * 4);

  v8f acc[4] = {};
  for (int khw = 0; khw < 4; ++khw) {
    const _Float16* Arow = xh + (size_t)(b * NTOK + nin[khw]) * DIMC;
    const int kb = khw * DIMC;
    for (int kc = 0; kc < DIMC; kc += 32) {
      const v16h a = *(const v16h*)(Arow + kc + bbase);
      const int k0 = kb + kc;
      const v16h b0 = *(const v16h*)(W0 + k0);
      const v16h b1 = *(const v16h*)(W1 + k0);
      const v16h b2 = *(const v16h*)(W2 + k0);
      const v16h b3 = *(const v16h*)(W3 + k0);
      acc[0] = __builtin_amdgcn_wmma_f32_16x16x32_f16(false, a, false, b0, (short)0, acc[0], false, false);
      acc[1] = __builtin_amdgcn_wmma_f32_16x16x32_f16(false, a, false, b1, (short)0, acc[1], false, false);
      acc[2] = __builtin_amdgcn_wmma_f32_16x16x32_f16(false, a, false, b2, (short)0, acc[2], false, false);
      acc[3] = __builtin_amdgcn_wmma_f32_16x16x32_f16(false, a, false, b3, (short)0, acc[3], false, false);
    }
  }

  const int r0 = (lane >> 4) << 3;
  const int cl = lane & 15;
#pragma unroll
  for (int t = 0; t < 4; ++t) {
    const int col = tn + (t << 4) + cl;
    const float bv = bias[col];
#pragma unroll
    for (int r = 0; r < 8; ++r)
      out[(size_t)(tm + r0 + r) * DIMC + col] = acc[t][r] + bv;
  }
}

// ---------------------------------------------------------------------------
// Row LayerNorm over 768 channels -> f16 output (kv GEMM operand).
// ---------------------------------------------------------------------------
__global__ void layernorm_kernel(const float* __restrict__ in,
                                 const float* __restrict__ g,
                                 const float* __restrict__ be,
                                 _Float16* __restrict__ out) {
  __shared__ float sm[64];
  const int row = blockIdx.x;
  const float* r = in + (size_t)row * DIMC;
  float s = 0.f, s2 = 0.f;
  for (int i = threadIdx.x; i < DIMC; i += 256) {
    const float v = r[i];
    s += v; s2 += v * v;
  }
#pragma unroll
  for (int o = 16; o > 0; o >>= 1) {
    s  += __shfl_xor(s,  o, 32);
    s2 += __shfl_xor(s2, o, 32);
  }
  const int wid = threadIdx.x >> 5, lane = threadIdx.x & 31;
  if (lane == 0) { sm[wid] = s; sm[32 + wid] = s2; }
  __syncthreads();
  if (threadIdx.x == 0) {
    float ts = 0.f, ts2 = 0.f;
    for (int w = 0; w < 8; ++w) { ts += sm[w]; ts2 += sm[32 + w]; }
    sm[0] = ts; sm[32] = ts2;
  }
  __syncthreads();
  const float mean = sm[0] * (1.0f / DIMC);
  const float var  = sm[32] * (1.0f / DIMC) - mean * mean;
  const float inv  = rsqrtf(var + 1e-5f);
  for (int i = threadIdx.x; i < DIMC; i += 256)
    out[(size_t)row * DIMC + i] = (_Float16)((r[i] - mean) * inv * g[i] + be[i]);
}

// ---------------------------------------------------------------------------
// RoPE for q: reads q f32 (B,N,768), writes f16 [b][h][n][d].
// ---------------------------------------------------------------------------
__global__ void rope_q_kernel(const float* __restrict__ q,
                              _Float16* __restrict__ qh) {
  const int tid = blockIdx.x * blockDim.x + threadIdx.x;
  const int total = BATCH * NTOK * NHEAD * (HD / 2);
  if (tid >= total) return;
  const int j  = tid & 31;
  const int h  = (tid >> 5) % NHEAD;
  const int bn = (tid >> 5) / NHEAD;
  const int n  = bn & (NTOK - 1);
  const int b  = bn >> 12;

  const int f = j >> 1;
  const float freq = __expf(-0.5756462732f * (float)f);  // 10000^(-4f/64)
  const float pos  = (j & 1) ? (float)(n >> 6) : (float)(n & 63);
  float sn, cs;
  __sincosf(pos * freq, &sn, &cs);

  const float* qp = q + ((size_t)(b * NTOK + n)) * DIMC + h * HD + 2 * j;
  const float x0 = qp[0], x1 = qp[1];
  const size_t o = (((size_t)(b * NHEAD + h)) * NTOK + n) * HD + 2 * j;
  qh[o]     = (_Float16)(x0 * cs - x1 * sn);
  qh[o + 1] = (_Float16)(x0 * sn + x1 * cs);
}

// ---------------------------------------------------------------------------
// RoPE for k + pack v: kv f32 (B,1024,1536) -> k_h [b][h][m][d],
// v transposed v_t [b][h][d][m] (contiguous m for P@V B-fragments).
// ---------------------------------------------------------------------------
__global__ void rope_kv_kernel(const float* __restrict__ kv,
                               _Float16* __restrict__ kh,
                               _Float16* __restrict__ vt) {
  const int tid = blockIdx.x * blockDim.x + threadIdx.x;
  const int total = BATCH * NRED * NHEAD * (HD / 2);
  if (tid >= total) return;
  const int j  = tid & 31;
  const int h  = (tid >> 5) % NHEAD;
  const int bm = (tid >> 5) / NHEAD;
  const int m  = bm & (NRED - 1);
  const int b  = bm >> 10;
  const int bh = b * NHEAD + h;

  const int f = j >> 1;
  const float freq = __expf(-0.5756462732f * (float)f);
  const float pos  = (j & 1) ? (0.5f + 2.0f * (float)(m >> 6))
                             : (0.5f + 2.0f * (float)(m & 63));
  float sn, cs;
  __sincosf(pos * freq, &sn, &cs);

  const float* kp = kv + ((size_t)(b * NRED + m)) * (2 * DIMC) + h * HD + 2 * j;
  const float* vp = kp + DIMC;
  const float x0 = kp[0], x1 = kp[1];
  const size_t ok = (((size_t)bh) * NRED + m) * HD + 2 * j;
  kh[ok]     = (_Float16)(x0 * cs - x1 * sn);
  kh[ok + 1] = (_Float16)(x0 * sn + x1 * cs);
  const size_t ov = (((size_t)bh) * HD + 2 * j) * NRED + m;
  vt[ov]        = (_Float16)vp[0];
  vt[ov + NRED] = (_Float16)vp[1];
}

// ---------------------------------------------------------------------------
// Attention: one 4-wave block per (b, h, 16-row n-tile).
//   S = q kT (WMMA, b128 loads) -> f16 LDS; masked softmax; P overwrites S;
//   out = P V (WMMA, A via ds_load_b128, B via contiguous v16h from v_t).
// Writes f16 output (proj GEMM operand).
// ---------------------------------------------------------------------------
__global__ void attention_kernel(const _Float16* __restrict__ qh,
                                 const _Float16* __restrict__ kh,
                                 const _Float16* __restrict__ vt,
                                 _Float16* __restrict__ outh) {
  __shared__ __align__(32) _Float16 Sh[16 * NRED];   // 32 KB, reused for P

  const int bh   = blockIdx.x >> 8;    // 0..23
  const int tile = blockIdx.x & 255;
  const int b = bh / NHEAD;
  const int h = bh % NHEAD;
  const int tm = tile << 4;
  const int wave = threadIdx.x >> 5;
  const int lane = threadIdx.x & 31;

  const _Float16* qbase = qh + (size_t)bh * NTOK * HD;
  const _Float16* kbase = kh + (size_t)bh * NRED * HD;
  const _Float16* vbase = vt + (size_t)bh * HD * NRED;

  const int abase = (lane < 16) ? 0 : 8;
  const int bbase = (lane < 16) ? 0 : 16;

  // q A-fragments for this 16-row tile (two K-steps of 32 over d=0..63)
  v16h aq0, aq1;
  {
    const _Float16* qrow = qbase + (size_t)(tm + (lane & 15)) * HD;
    aq0 = *(const v16h*)(qrow + bbase);
    aq1 = *(const v16h*)(qrow + 32 + bbase);
  }

  // S = q @ k^T : each wave produces 16 column tiles of 16
  for (int ct = wave * 16; ct < wave * 16 + 16; ++ct) {
    const int mcol = (ct << 4) + (lane & 15);
    const _Float16* krow = kbase + (size_t)mcol * HD;
    const v16h b0 = *(const v16h*)(krow + bbase);        // B[k=d, n=m] = k[m,d]
    const v16h b1 = *(const v16h*)(krow + 32 + bbase);
    v8f acc = {};
    acc = __builtin_amdgcn_wmma_f32_16x16x32_f16(false, aq0, false, b0, (short)0, acc, false, false);
    acc = __builtin_amdgcn_wmma_f32_16x16x32_f16(false, aq1, false, b1, (short)0, acc, false, false);
    const int r0 = (lane >> 4) << 3;
    const int cc = (ct << 4) + (lane & 15);
#pragma unroll
    for (int r = 0; r < 8; ++r)
      Sh[(r0 + r) * NRED + cc] = (_Float16)acc[r];
  }
  __syncthreads();

  // Masked softmax: thread t -> row = t>>4, handles cols (t&15) + 16*i.
  {
    const int row = threadIdx.x >> 4;
    const int sub = threadIdx.x & 15;
    const int n = tm + row;
    const int xpos = n >> 7;
    const int ox = n >> 6;
    const int oy = n & 63;
    const int ypos = (ox + oy * 64) >> 7;
    const int vis = xpos * 2 + ypos + 1;

    float vals[64];
    float mx = -3.0e30f;
#pragma unroll
    for (int i = 0; i < 64; ++i) {
      const int m = sub + (i << 4);
      const float sv = (m < vis) ? (float)Sh[row * NRED + m] * 0.125f : -3.0e30f;
      vals[i] = sv;
      mx = fmaxf(mx, sv);
    }
#pragma unroll
    for (int o = 8; o > 0; o >>= 1) mx = fmaxf(mx, __shfl_xor(mx, o, 16));
    float sum = 0.f;
#pragma unroll
    for (int i = 0; i < 64; ++i) {
      const float e = __expf(vals[i] - mx);
      vals[i] = e;
      sum += e;
    }
#pragma unroll
    for (int o = 8; o > 0; o >>= 1) sum += __shfl_xor(sum, o, 16);
    const float inv = 1.0f / sum;
#pragma unroll
    for (int i = 0; i < 64; ++i) {
      const int m = sub + (i << 4);
      Sh[row * NRED + m] = (_Float16)(vals[i] * inv);   // same slots this thread read
    }
  }
  __syncthreads();

  // out = P @ V : wave w handles the 16-wide d-tile dt = w*16
  {
    const int dt = wave << 4;
    const int dcol = dt + (lane & 15);
    const int prow = lane & 15;
    const _Float16* Vrow = vbase + (size_t)dcol * NRED + bbase;  // B[k=m, n=d] = vT[d,m]
    v8f acc = {};
    for (int k0 = 0; k0 < NRED; k0 += 32) {
      const _Float16* Pr = &Sh[prow * NRED + k0];
      const v16h a = *(const v16h*)(Pr + bbase);
      const v16h bv = *(const v16h*)(Vrow + k0);
      acc = __builtin_amdgcn_wmma_f32_16x16x32_f16(false, a, false, bv, (short)0, acc, false, false);
    }
    const int r0  = (lane >> 4) << 3;
    const int col = lane & 15;
#pragma unroll
    for (int r = 0; r < 8; ++r) {
      const int n = tm + r0 + r;
      outh[((size_t)b * NTOK + n) * DIMC + h * HD + dt + col] = (_Float16)acc[r];
    }
  }
}

// ---------------------------------------------------------------------------
extern "C" void kernel_launch(void* const* d_in, const int* in_sizes, int n_in,
                              void* d_out, int out_size, void* d_ws, size_t ws_size,
                              hipStream_t stream) {
  const float* x      = (const float*)d_in[0];
  const float* Wq     = (const float*)d_in[1];
  const float* Wkv    = (const float*)d_in[2];
  const float* sr_w   = (const float*)d_in[3];
  const float* sr_b   = (const float*)d_in[4];
  const float* ln_g   = (const float*)d_in[5];
  const float* ln_b   = (const float*)d_in[6];
  const float* proj_w = (const float*)d_in[7];
  const float* proj_b = (const float*)d_in[8];
  float* out = (float*)d_out;

  char* ws = (char*)d_ws;
  _Float16* x_h    = (_Float16*)(ws);                    // 8192*768*2  = 12582912
  _Float16* wq_h   = (_Float16*)(ws + 12582912);         //  768*768*2  =  1179648
  _Float16* wkv_h  = (_Float16*)(ws + 13762560);         // 1536*768*2  =  2359296
  _Float16* wpr_h  = (_Float16*)(ws + 16121856);         //  768*768*2  =  1179648
  _Float16* wsr_h  = (_Float16*)(ws + 17301504);         //  768*3072*2 =  4718592
  float*    q_tmp  = (float*)(ws + 22020096);            // 8192*768*4  = 25165824
  _Float16* attn_h = (_Float16*)(ws + 22020096);         // reuses q_tmp region
  _Float16* q_h    = (_Float16*)(ws + 47185920);         // 8192*768*2  = 12582912
  float*    xr     = (float*)(ws + 59768832);            // 2048*768*4  =  6291456
  _Float16* xrln_h = (_Float16*)(ws + 66060288);         // 2048*768*2  =  3145728
  float*    kv     = (float*)(ws + 69206016);            // 2048*1536*4 = 12582912
  _Float16* k_h    = (_Float16*)(ws + 81788928);         // 2048*768*2  =  3145728
  _Float16* v_t    = (_Float16*)(ws + 84934656);         // 2048*768*2  =  3145728

  const int M_q = BATCH * NTOK;   // 8192
  const int M_r = BATCH * NRED;   // 2048

  // 0) convert operands to f16 once
  cvt_f16_kernel<<<(M_q * DIMC + 255) / 256, 256, 0, stream>>>(x, x_h, M_q * DIMC);
  cvt_f16_kernel<<<(DIMC * DIMC + 255) / 256, 256, 0, stream>>>(Wq, wq_h, DIMC * DIMC);
  cvt_f16_kernel<<<(2 * DIMC * DIMC + 255) / 256, 256, 0, stream>>>(Wkv, wkv_h, 2 * DIMC * DIMC);
  cvt_f16_kernel<<<(DIMC * DIMC + 255) / 256, 256, 0, stream>>>(proj_w, wpr_h, DIMC * DIMC);
  cvt_srw_kernel<<<(DIMC * DIMC * 4 + 255) / 256, 256, 0, stream>>>(sr_w, wsr_h);

  // 1) q = x @ Wq^T  (f32 for RoPE precision)
  gemm_h_kernel<<<(M_q / 64) * (DIMC / 64), 128, 0, stream>>>(
      x_h, wq_h, nullptr, q_tmp, M_q, DIMC, DIMC);

  // 2) RoPE(q) -> f16 [b][h][n][d]
  {
    const int total = BATCH * NTOK * NHEAD * (HD / 2);
    rope_q_kernel<<<(total + 255) / 256, 256, 0, stream>>>(q_tmp, q_h);
  }

  // 3) stride-2 conv as GEMM (+ bias), pixel-major K
  conv_h_kernel<<<(M_r / 64) * (DIMC / 64), 128, 0, stream>>>(x_h, wsr_h, sr_b, xr);

  // 4) LayerNorm rows -> f16
  layernorm_kernel<<<M_r, 256, 0, stream>>>(xr, ln_g, ln_b, xrln_h);

  // 5) kv = xr_ln @ Wkv^T
  gemm_h_kernel<<<(M_r / 64) * ((2 * DIMC) / 64), 128, 0, stream>>>(
      xrln_h, wkv_h, nullptr, kv, M_r, 2 * DIMC, DIMC);

  // 6) RoPE(k) + pack v (transposed) -> f16
  {
    const int total = BATCH * NRED * NHEAD * (HD / 2);
    rope_kv_kernel<<<(total + 255) / 256, 256, 0, stream>>>(kv, k_h, v_t);
  }

  // 7) attention -> f16 attn_h (reuses q_tmp region; q_tmp consumed in step 2)
  attention_kernel<<<BATCH * NHEAD * (NTOK / 16), 128, 0, stream>>>(
      q_h, k_h, v_t, attn_h);

  // 8) out = attn @ proj_w^T + proj_b  (f32 output)
  gemm_h_kernel<<<(M_q / 64) * (DIMC / 64), 128, 0, stream>>>(
      attn_h, wpr_h, proj_b, out, M_q, DIMC, DIMC);
}
